// MultiHeadAttention_25537875542283
// MI455X (gfx1250) — compile-verified, matrix operand order checked by reading, not run on hardware
//
#include <hip/hip_runtime.h>
#include <hip/hip_bf16.h>

#define N_EMBD 384
#define N_HEADS 6
#define HEAD 64
#define BSZ 128
#define TLEN 256

typedef __attribute__((ext_vector_type(16))) __bf16 v16bf;
typedef __attribute__((ext_vector_type(8)))  __bf16 v8bf;
typedef __attribute__((ext_vector_type(8)))  float  v8f;
typedef __attribute__((ext_vector_type(4)))  unsigned int v4u;
typedef __attribute__((ext_vector_type(8)))  int    v8i;
typedef __attribute__((ext_vector_type(4)))  int    v4i;

static __device__ inline v16bf cat16(v8bf lo, v8bf hi) {
  return __builtin_shufflevector(lo, hi, 0,1,2,3,4,5,6,7,8,9,10,11,12,13,14,15);
}
static __device__ inline v8f wmma_bf16(v16bf a, v16bf b, v8f c) {
  return __builtin_amdgcn_wmma_f32_16x16x32_bf16(false, a, false, b, (short)0, c, false, false);
}
static __device__ inline v8bf ld8(const __bf16* p) { return *(const v8bf*)p; }

#if __has_builtin(__builtin_amdgcn_tensor_load_to_lds)
#define HAVE_TDM 1
#else
#define HAVE_TDM 0
#endif

#if HAVE_TDM
// Issue one TDM descriptor copying (rows x 2048) contiguous bf16 elements
// global -> LDS.  data_size=2B, tile == tensor, stride 2048.
static __device__ inline void tdm_issue_copy(__bf16* dst, const __bf16* src,
                                             int rows) {
  const unsigned int ldsoff = (unsigned int)(uintptr_t)dst;  // low 32b = LDS off
  const unsigned long long ga = (unsigned long long)(uintptr_t)src;
  v4u g0 = { 1u, ldsoff, (unsigned int)ga,
             (unsigned int)((ga >> 32) & 0x1FFFFFFu) | (2u << 30) };
  v8i g1 = { (int)(1u << 16),                 // [17:16] data_size=1 -> 2 bytes
             (int)(2048u << 16),              // [63:48] tensor_dim0 lo16
             (int)((unsigned int)rows << 16), // [95:80] tensor_dim1 lo16
             (int)(2048u << 16),              // [127:112] tile_dim0
             rows,                            // [143:128] tile_dim1
             2048,                            // [191:160] tensor_dim0_stride
             0, 0 };
  v4i g2 = {0,0,0,0}, g3 = {0,0,0,0};
#if __clang_major__ >= 23
  v8i g4 = {0,0,0,0,0,0,0,0};
  __builtin_amdgcn_tensor_load_to_lds(g0, g1, g2, g3, g4, 0);
#else
  __builtin_amdgcn_tensor_load_to_lds(g0, g1, g2, g3, 0);
#endif
}
#endif

// Stage rows*2048 contiguous bf16 elements into LDS, whole block synchronized.
static __device__ inline void stage_lds(__bf16* dst, const __bf16* src,
                                        int rows, int tid) {
#if HAVE_TDM
  if (tid < 32) {
    tdm_issue_copy(dst, src, rows);
    __builtin_amdgcn_s_wait_tensorcnt(0);
  }
  __syncthreads();
#else
  for (int i = tid; i < rows * 256; i += 256)
    ((v8bf*)dst)[i] = ((const v8bf*)src)[i];
  __syncthreads();
#endif
}

// ---------------------------------------------------------------------------
// Kernel 0: one-shot bf16 conversion.
//  xb[r][c]            = x[r][c]
//  wqkvt[mm][h][d][c]  = W{q,k,v}[h][c][d]   (B-operand friendly transpose)
//  wot[g][d][c]        = Wo[c][g*64+d]
// ---------------------------------------------------------------------------
__global__ __launch_bounds__(256) void cvt_kernel(
    const float* __restrict__ x,
    const float* __restrict__ Wq, const float* __restrict__ Wk,
    const float* __restrict__ Wv, const float* __restrict__ Wo,
    __bf16* __restrict__ xb, __bf16* __restrict__ wqkvt, __bf16* __restrict__ wot)
{
  const int NX = BSZ * TLEN * N_EMBD;
  const int NW = 3 * N_HEADS * HEAD * N_EMBD;
  const int NO = N_HEADS * HEAD * N_EMBD;
  for (int i = blockIdx.x * 256 + threadIdx.x; i < NX + NW + NO;
       i += gridDim.x * 256) {
    if (i < NX) {
      xb[i] = (__bf16)x[i];
    } else if (i < NX + NW) {
      const int j = i - NX;
      const int mm = j / (N_HEADS * HEAD * N_EMBD);
      const int r  = j % (N_HEADS * HEAD * N_EMBD);
      const int h = r / (HEAD * N_EMBD);
      const int q = r % (HEAD * N_EMBD);
      const int d = q / N_EMBD, c = q % N_EMBD;
      const float* W = (mm == 0 ? Wq : (mm == 1 ? Wk : Wv));
      wqkvt[j] = (__bf16)W[((size_t)h * N_EMBD + c) * HEAD + d];
    } else {
      const int j = i - NX - NW;
      const int g = j / (HEAD * N_EMBD);
      const int q = j % (HEAD * N_EMBD);
      const int d = q / N_EMBD, c = q % N_EMBD;
      wot[j] = (__bf16)Wo[(size_t)c * N_EMBD + g * HEAD + d];
    }
  }
}

// ---------------------------------------------------------------------------
// Kernel 1: Q/K/V projection.  grid = (256 row-tiles, 6 heads, 3 matrices)
// Q,K stored bf16 [b*H+h][t][d]; V stored transposed bf16 [b*H+h][d][t].
// Q pre-scaled by N_EMBD^-0.5.
// ---------------------------------------------------------------------------
__global__ __launch_bounds__(256) void qkv_kernel(
    const __bf16* __restrict__ xb, const __bf16* __restrict__ wqkvt,
    const float* __restrict__ bq, const float* __restrict__ bk,
    const float* __restrict__ bv,
    __bf16* __restrict__ qb, __bf16* __restrict__ kbuf, __bf16* __restrict__ vt)
{
  __shared__ __bf16 WT[HEAD][N_EMBD];          // 48 KB: WT[d][c]
  const int h  = blockIdx.y;
  const int mm = blockIdx.z;                   // 0=Q 1=K 2=V
  stage_lds(&WT[0][0], wqkvt + (size_t)(mm * N_HEADS + h) * HEAD * N_EMBD,
            12, threadIdx.x);
  const float* bias = (mm == 0 ? bq : (mm == 1 ? bk : bv)) + h * HEAD;

  const int wave = threadIdx.x >> 5;
  const int lane = threadIdx.x & 31;
  const int lane16 = lane & 15;
  const int hi = lane >> 4;

  const int r0 = (blockIdx.x * 8 + wave) * 16;
  const int arow = r0 + lane16;

  v8f acc[4] = {};
  for (int kk = 0; kk < 12; ++kk) {            // K = 384 in chunks of 32
    const int kb0 = kk * 32 + hi * 8;          // 16-bit A layout K-halves
    const int kb1 = kk * 32 + 16 + hi * 8;
    v16bf a = cat16(ld8(xb + (size_t)arow * N_EMBD + kb0),
                    ld8(xb + (size_t)arow * N_EMBD + kb1));
    #pragma unroll
    for (int nt = 0; nt < 4; ++nt) {
      const int d = nt * 16 + lane16;
      v16bf bmat = cat16(ld8(&WT[d][kb0]), ld8(&WT[d][kb1]));
      acc[nt] = wmma_bf16(a, bmat, acc[nt]);
    }
  }

  const int bidx  = r0 >> 8;                   // batch (T = 256)
  const int tbase = r0 & (TLEN - 1);
  const float qscale = rsqrtf((float)N_EMBD);
  #pragma unroll
  for (int nt = 0; nt < 4; ++nt) {
    const int d = nt * 16 + lane16;
    const float bval = bias[d];
    #pragma unroll
    for (int i = 0; i < 8; ++i) {
      const int t = tbase + i + hi * 8;        // C layout: lane half -> M+8
      float val = acc[nt][i] + bval;
      if (mm == 0) val *= qscale;
      if (mm == 2) {
        vt[((size_t)(bidx * N_HEADS + h) * HEAD + d) * TLEN + t] = (__bf16)val;
      } else {
        __bf16* dst = (mm == 0 ? qb : kbuf);
        dst[((size_t)(bidx * N_HEADS + h) * TLEN + t) * HEAD + d] = (__bf16)val;
      }
    }
  }
}

// ---------------------------------------------------------------------------
// Kernel 2: flash attention per (b,h).  Reference: S = K @ Q^T * s, tril mask,
// softmax over last dim, O = P @ V.  Rows = key position t, reduce over s.
// Q (32 KB) and V^T (32 KB) panels staged once per block via TDM; the inner
// loop is then pure ds_load_b128 -> v_wmma.
// ---------------------------------------------------------------------------
__global__ __launch_bounds__(256) void attn_kernel(
    const __bf16* __restrict__ qb, const __bf16* __restrict__ kbuf,
    const __bf16* __restrict__ vt, __bf16* __restrict__ ab)
{
  __shared__ __bf16 Qs[TLEN][HEAD];            // 32 KB
  __shared__ __bf16 Vs[HEAD][TLEN];            // 32 KB (V transposed: [d][s])
  __shared__ __bf16 P[8][16][32];              // 8 KB per-wave P transpose
  const int bh = blockIdx.x;
  const int b = bh / N_HEADS, h = bh % N_HEADS;
  const __bf16* Qp = qb   + (size_t)bh * TLEN * HEAD;
  const __bf16* Kp = kbuf + (size_t)bh * TLEN * HEAD;
  const __bf16* Vp = vt   + (size_t)bh * HEAD * TLEN;

#if HAVE_TDM
  if (threadIdx.x < 32) {                      // one wave, two descriptors
    tdm_issue_copy(&Qs[0][0], Qp, 8);
    tdm_issue_copy(&Vs[0][0], Vp, 8);
    __builtin_amdgcn_s_wait_tensorcnt(0);
  }
  __syncthreads();
#else
  for (int i = threadIdx.x; i < (TLEN * HEAD) / 8; i += 256) {
    ((v8bf*)&Qs[0][0])[i] = ((const v8bf*)Qp)[i];
    ((v8bf*)&Vs[0][0])[i] = ((const v8bf*)Vp)[i];
  }
  __syncthreads();
#endif

  const int wave = threadIdx.x >> 5;
  const int lane = threadIdx.x & 31;
  const int lane16 = lane & 15;
  const int hi = lane >> 4;

  for (int tt = wave; tt < 16; tt += 8) {
    const int t0 = tt * 16;
    v16bf aK[2];                               // K rows, reused across s
    #pragma unroll
    for (int kk = 0; kk < 2; ++kk) {
      const int kb0 = kk * 32 + hi * 8, kb1 = kk * 32 + 16 + hi * 8;
      aK[kk] = cat16(ld8(Kp + (size_t)(t0 + lane16) * HEAD + kb0),
                     ld8(Kp + (size_t)(t0 + lane16) * HEAD + kb1));
    }
    v8f o[4] = {};
    float m[8], l[8];
    #pragma unroll
    for (int i = 0; i < 8; ++i) { m[i] = -1e30f; l[i] = 0.f; }

    const int nsb = t0 / 32 + 1;               // causal: s <= t0+15
    for (int sb = 0; sb < nsb; ++sb) {
      const int s0 = sb * 32;
      v8f sacc[2] = {};
      #pragma unroll
      for (int nt = 0; nt < 2; ++nt) {
        const int s = s0 + nt * 16 + lane16;
        #pragma unroll
        for (int kk = 0; kk < 2; ++kk) {
          const int kb0 = kk * 32 + hi * 8, kb1 = kk * 32 + 16 + hi * 8;
          v16bf bq_ = cat16(ld8(&Qs[s][kb0]), ld8(&Qs[s][kb1]));
          sacc[nt] = wmma_bf16(aK[kk], bq_, sacc[nt]);
        }
      }
      // mask + online softmax (C layout: lane = col s, VGPR i = row t or t+8)
      #pragma unroll
      for (int i = 0; i < 8; ++i) {
        const int trow = t0 + i + hi * 8;
        #pragma unroll
        for (int nt = 0; nt < 2; ++nt) {
          const int s = s0 + nt * 16 + lane16;
          if (s > trow) sacc[nt][i] = -1e30f;
        }
        float rmax = fmaxf(sacc[0][i], sacc[1][i]);
        #pragma unroll
        for (int off = 8; off >= 1; off >>= 1)
          rmax = fmaxf(rmax, __shfl_xor(rmax, off, 16));
        const float mn = fmaxf(m[i], rmax);
        const float sc = __expf(m[i] - mn);
        const float p0 = __expf(sacc[0][i] - mn);
        const float p1 = __expf(sacc[1][i] - mn);
        float rsum = p0 + p1;
        #pragma unroll
        for (int off = 8; off >= 1; off >>= 1)
          rsum += __shfl_xor(rsum, off, 16);
        m[i] = mn;
        l[i] = l[i] * sc + rsum;
        #pragma unroll
        for (int nt = 0; nt < 4; ++nt) o[nt][i] *= sc;
        P[wave][i + hi * 8][lane16]      = (__bf16)p0;
        P[wave][i + hi * 8][16 + lane16] = (__bf16)p1;
      }
      asm volatile("s_wait_dscnt 0" ::: "memory");     // LDS stores visible
      v16bf pa = cat16(ld8(&P[wave][lane16][hi * 8]),
                       ld8(&P[wave][lane16][16 + hi * 8]));
      #pragma unroll
      for (int nt = 0; nt < 4; ++nt) {
        const int d = nt * 16 + lane16;
        v16bf bv_ = cat16(ld8(&Vs[d][s0 + hi * 8]),
                          ld8(&Vs[d][s0 + 16 + hi * 8]));
        o[nt] = wmma_bf16(pa, bv_, o[nt]);
      }
    }
    #pragma unroll
    for (int i = 0; i < 8; ++i) {
      const float inv = 1.f / l[i];
      const int t = t0 + i + hi * 8;
      #pragma unroll
      for (int nt = 0; nt < 4; ++nt)
        ab[(size_t)(b * TLEN + t) * N_EMBD + h * HEAD + nt * 16 + lane16] =
            (__bf16)(o[nt][i] * inv);
    }
  }
}

// ---------------------------------------------------------------------------
// Kernel 3: out = attn @ Wo + bo.  grid = (256 row-tiles, 6 col-groups of 64)
// ---------------------------------------------------------------------------
__global__ __launch_bounds__(256) void proj_kernel(
    const __bf16* __restrict__ ab, const __bf16* __restrict__ wot,
    const float* __restrict__ bo, float* __restrict__ out)
{
  __shared__ __bf16 WT[HEAD][N_EMBD];
  const int g = blockIdx.y;
  stage_lds(&WT[0][0], wot + (size_t)g * HEAD * N_EMBD, 12, threadIdx.x);

  const int wave = threadIdx.x >> 5;
  const int lane = threadIdx.x & 31;
  const int lane16 = lane & 15;
  const int hi = lane >> 4;
  const int r0 = (blockIdx.x * 8 + wave) * 16;
  const int arow = r0 + lane16;

  v8f acc[4] = {};
  for (int kk = 0; kk < 12; ++kk) {
    const int kb0 = kk * 32 + hi * 8, kb1 = kk * 32 + 16 + hi * 8;
    v16bf a = cat16(ld8(ab + (size_t)arow * N_EMBD + kb0),
                    ld8(ab + (size_t)arow * N_EMBD + kb1));
    #pragma unroll
    for (int nt = 0; nt < 4; ++nt) {
      const int d = nt * 16 + lane16;
      v16bf bmat = cat16(ld8(&WT[d][kb0]), ld8(&WT[d][kb1]));
      acc[nt] = wmma_bf16(a, bmat, acc[nt]);
    }
  }
  #pragma unroll
  for (int nt = 0; nt < 4; ++nt) {
    const int col = g * HEAD + nt * 16 + lane16;
    const float bb = bo[col];
    #pragma unroll
    for (int i = 0; i < 8; ++i) {
      const int r = r0 + i + hi * 8;
      out[(size_t)r * N_EMBD + col] = acc[nt][i] + bb;
    }
  }
}

extern "C" void kernel_launch(void* const* d_in, const int* in_sizes, int n_in,
                              void* d_out, int out_size, void* d_ws, size_t ws_size,
                              hipStream_t stream) {
  const float* x  = (const float*)d_in[0];
  const float* Wq = (const float*)d_in[1];
  const float* Wk = (const float*)d_in[2];
  const float* Wv = (const float*)d_in[3];
  const float* bq = (const float*)d_in[4];
  const float* bk = (const float*)d_in[5];
  const float* bv = (const float*)d_in[6];
  const float* Wo = (const float*)d_in[7];
  const float* bo = (const float*)d_in[8];
  float* out = (float*)d_out;

  const size_t QS = (size_t)N_HEADS * BSZ * TLEN * HEAD;   // 12.58M elems
  const size_t NW = (size_t)3 * N_HEADS * HEAD * N_EMBD;   // 442,368
  __bf16* qb    = (__bf16*)d_ws;
  __bf16* kb    = qb + QS;
  __bf16* vt    = kb + QS;
  __bf16* xb    = vt + QS;          // [B*T][384] bf16 (aliased with ab below)
  __bf16* ab    = xb;               // xb dead once attn runs; reuse as attn out
  __bf16* wqkvt = xb + QS;
  __bf16* wot   = wqkvt + NW;

  cvt_kernel<<<dim3(1024), 256, 0, stream>>>(x, Wq, Wk, Wv, Wo, xb, wqkvt, wot);
  dim3 gA(BSZ * TLEN / 128, N_HEADS, 3);                    // 256 x 6 x 3
  qkv_kernel<<<gA, 256, 0, stream>>>(xb, wqkvt, bq, bk, bv, qb, kb, vt);
  attn_kernel<<<dim3(BSZ * N_HEADS), 256, 0, stream>>>(qb, kb, vt, ab);
  proj_kernel<<<dim3(BSZ * TLEN / 128, N_EMBD / HEAD), 256, 0, stream>>>(ab, wot, bo, out);
}
